// MultiHeadAttention_21526376087829
// MI455X (gfx1250) — compile-verified, matrix operand order checked by reading, not run on hardware
//
#include <hip/hip_runtime.h>

typedef __bf16 bf16;
typedef __attribute__((ext_vector_type(16))) __bf16 v16bf;
typedef __attribute__((ext_vector_type(8)))  __bf16 bf16x8;
typedef __attribute__((ext_vector_type(8)))  float  v8f;

static constexpr int B_ = 2, T_ = 2048, C_ = 2048, NH_ = 16, HD_ = 128;

union FragU { v16bf v; bf16x8 h[2]; };

// Load a 16x32 bf16 WMMA fragment (A layout; B layout is the lane-symmetric
// twin, so the same loader serves B when 'base' is the [N,K] row-major matrix).
// Lane L: row = row0 + (L&15); K-chunks at k0+(L>>4)*8 and k0+16+(L>>4)*8.
__device__ __forceinline__ v16bf load_frag(const bf16* __restrict__ base,
                                           long row0, int ld, int k0, int lane) {
  const int r  = lane & 15;
  const int ko = (lane >> 4) << 3;
  const bf16* p = base + (row0 + r) * (long)ld + k0 + ko;
  FragU u;
  u.h[0] = *(const bf16x8*)(p);
  u.h[1] = *(const bf16x8*)(p + 16);
  return u.v;
}

__device__ __forceinline__ v8f wmma_bf16(v16bf a, v16bf b, v8f c) {
  return __builtin_amdgcn_wmma_f32_16x16x32_bf16(false, a, false, b,
                                                 (short)0, c, false, false);
}

// ---- gfx1250 async copy: global -> LDS, 16B per lane, tracked by ASYNCcnt.
// VDST holds the LDS byte address (low 32 bits of the flat shared pointer).
__device__ __forceinline__ void async_copy_b128(unsigned lds_byte,
                                                const bf16* g) {
  asm volatile("global_load_async_to_lds_b128 %0, %1, off"
               :: "v"(lds_byte), "v"((unsigned long long)(size_t)g)
               : "memory");
}
__device__ __forceinline__ void wait_async_le4() {
  asm volatile("s_wait_asynccnt 0x4" ::: "memory");
}
__device__ __forceinline__ void wait_async_le32() {
  asm volatile("s_wait_asynccnt 0x20" ::: "memory");
}
__device__ __forceinline__ void wait_async_le0() {
  asm volatile("s_wait_asynccnt 0x0" ::: "memory");
}
__device__ __forceinline__ void wait_ds0() {
  asm volatile("s_wait_dscnt 0x0" ::: "memory");
}

// ---------------------------------------------------------------- convert
__global__ void __launch_bounds__(256)
cvt_f32_bf16(const float* __restrict__ in, bf16* __restrict__ out, long n) {
  long i = (long)blockIdx.x * 256 + threadIdx.x;
  if (i < n) out[i] = (bf16)in[i];
}

// ---------------------------------------------------------------- tiled GEMM
// C = A * W^T.  A:[M,K] bf16 row-major, W:[N,K] bf16 row-major.
// Block tile 128x128, BK=32, 256 threads (8 waves), async double-buffered LDS.
// Wave w: rows (w>>1)*32..+32, cols (w&1)*64..+64  -> 8 WMMA / k-step.
template <typename OutT>
__global__ void __launch_bounds__(256, 1)
gemm_tiled(const bf16* __restrict__ A, const bf16* __restrict__ W,
           OutT* __restrict__ Cm, int N, int K) {
  __shared__ __align__(16) bf16 As[2][128 * 32];
  __shared__ __align__(16) bf16 Bs[2][128 * 32];

  const int tid  = threadIdx.x;
  const int lane = tid & 31;
  const int w    = tid >> 5;
  const int rw   = (w >> 1) * 32;   // wave row offset in tile
  const int cw   = (w & 1) * 64;    // wave col offset in tile
  const long mtile = (long)blockIdx.y * 128;
  const long ntile = (long)blockIdx.x * 128;

  const unsigned asbase = (unsigned)(size_t)(&As[0][0]);
  const unsigned bsbase = (unsigned)(size_t)(&Bs[0][0]);

  // each thread stages 2 A-chunks + 2 B-chunks (16B each) per k-stage
  auto stage_load = [&](int stg, int kt) {
    const unsigned soff = (unsigned)stg * (128 * 32 * 2);
#pragma unroll
    for (int i = 0; i < 2; ++i) {
      const int c   = tid + i * 256;       // 0..511
      const int row = c >> 2;              // 0..127
      const int c8  = (c & 3) * 8;         // 0,8,16,24
      const unsigned loff = soff + (unsigned)(row * 32 + c8) * 2;
      async_copy_b128(asbase + loff, A + (mtile + row) * (long)K + kt + c8);
      async_copy_b128(bsbase + loff, W + (ntile + row) * (long)K + kt + c8);
    }
  };

  v8f acc0[4], acc1[4];
#pragma unroll
  for (int j = 0; j < 4; ++j) {
    acc0[j] = (v8f){0.f,0.f,0.f,0.f,0.f,0.f,0.f,0.f};
    acc1[j] = (v8f){0.f,0.f,0.f,0.f,0.f,0.f,0.f,0.f};
  }

  stage_load(0, 0);
  int s = 0;
  for (int kt = 0; kt < K; kt += 32) {
    const bool has_next = (kt + 32) < K;
    if (has_next) {
      stage_load(s ^ 1, kt + 32);
      wait_async_le4();        // current stage (issued first) has retired
    } else {
      wait_async_le0();
    }
    __syncthreads();           // every wave has waited its own async ops

    const bf16* as = &As[s][0];
    const bf16* bs = &Bs[s][0];
    v16bf a0 = load_frag(as, rw,      32, 0, lane);
    v16bf a1 = load_frag(as, rw + 16, 32, 0, lane);
#pragma unroll
    for (int j = 0; j < 4; ++j) {
      v16bf bfr = load_frag(bs, cw + j * 16, 32, 0, lane);
      acc0[j] = wmma_bf16(a0, bfr, acc0[j]);
      acc1[j] = wmma_bf16(a1, bfr, acc1[j]);
    }
    __syncthreads();           // stage s readable until here; safe to refill next iter
    s ^= 1;
  }

  const int m0 = (lane >> 4) * 8, n0 = lane & 15;
#pragma unroll
  for (int j = 0; j < 4; ++j)
#pragma unroll
    for (int r = 0; r < 8; ++r) {
      Cm[(mtile + rw + m0 + r) * (long)N + ntile + cw + j * 16 + n0] =
          (OutT)acc0[j][r];
      Cm[(mtile + rw + 16 + m0 + r) * (long)N + ntile + cw + j * 16 + n0] =
          (OutT)acc1[j][r];
    }
}

// ---------------------------------------------------------------- RoPE + split
// qkv: [B*T, 3C] bf16.  Writes q,k roped as [B,NH,T,HD]; v transposed [B,NH,HD,T].
__global__ void __launch_bounds__(64)
rope_split(const bf16* __restrict__ qkv,
           const float* __restrict__ cosp, const float* __restrict__ sinp,
           bf16* __restrict__ qo, bf16* __restrict__ ko, bf16* __restrict__ vo) {
  const int gid = blockIdx.x;                 // (b*T + t)*NH + h
  const int h = gid & (NH_ - 1);
  const int t = (gid >> 4) & (T_ - 1);
  const int b = gid >> 15;                    // T_*NH_ = 32768
  const int d = threadIdx.x;                  // 0..63  (HD/2)
  const long row = (long)(b * T_ + t) * (3 * C_);
  const int base = h * HD_;
  float q0 = (float)qkv[row + base + d];
  float q1 = (float)qkv[row + base + 64 + d];
  float k0 = (float)qkv[row + C_ + base + d];
  float k1 = (float)qkv[row + C_ + base + 64 + d];
  float v0 = (float)qkv[row + 2 * C_ + base + d];
  float v1 = (float)qkv[row + 2 * C_ + base + 64 + d];
  float cv = cosp[t * 64 + d];
  float sv = sinp[t * 64 + d];
  const long qb = ((long)(b * NH_ + h) * T_ + t) * HD_;
  qo[qb + d]      = (bf16)(q0 * cv - q1 * sv);
  qo[qb + 64 + d] = (bf16)(q0 * sv + q1 * cv);
  ko[qb + d]      = (bf16)(k0 * cv - k1 * sv);
  ko[qb + 64 + d] = (bf16)(k0 * sv + k1 * cv);
  const long vb = (long)(b * NH_ + h) * HD_ * T_;
  vo[vb + (long)d * T_ + t]        = (bf16)v0;
  vo[vb + (long)(d + 64) * T_ + t] = (bf16)v1;
}

// ---------------------------------------------------------------- flash attention
// One wave (32 threads) per 16-row Q tile of one (b,h). Output [B,T,C] bf16.
// K/V tiles are async-copied to LDS double-buffered: while tile kk is consumed
// (8 QK WMMAs + softmax + 8 PV WMMAs), tile kk+32 streams in via ASYNCcnt DMA.
__global__ void __launch_bounds__(32, 1)
attn_kernel(const bf16* __restrict__ q, const bf16* __restrict__ k,
            const bf16* __restrict__ vt, bf16* __restrict__ ao) {
  const int lane = threadIdx.x & 31;
  const int bid = blockIdx.x;
  const int qt = bid & (T_ / 16 - 1);
  const int h  = (bid >> 7) & (NH_ - 1);
  const int b  = bid >> 11;
  const int qi0 = qt * 16;

  const bf16* qp = q  + (long)(b * NH_ + h) * T_ * HD_;
  const bf16* kp = k  + (long)(b * NH_ + h) * T_ * HD_;
  const bf16* vp = vt + (long)(b * NH_ + h) * HD_ * T_;

  __shared__ __align__(16) bf16  ksm[2][32 * 128];  // K tile: 32 keys x 128 feat
  __shared__ __align__(16) bf16  vsm[2][128 * 32];  // V^T tile: 128 hd x 32 keys
  __shared__ __align__(16) float ssm[16][34];       // raw scores, 32 keys + pad
  __shared__ __align__(16) bf16  psm[16][40];       // probs (ld=40 -> 16B rows)
  __shared__ float alphas[16];
  __shared__ float linv[16];

  const unsigned ksbase = (unsigned)(size_t)(&ksm[0][0]);
  const unsigned vsbase = (unsigned)(size_t)(&vsm[0][0]);

  // Stage K (32x128) + V^T (128x32) tiles for key range [kk2, kk2+32).
  // 1024 x 16B chunks / 32 lanes = 32 async copies per lane (ASYNCcnt +32).
  auto stage_kv = [&](int stg, int kk2) {
    const unsigned ko_ = ksbase + (unsigned)stg * (32 * 128 * 2);
    const unsigned vo_ = vsbase + (unsigned)stg * (128 * 32 * 2);
#pragma unroll
    for (int i = 0; i < 16; ++i) {
      const int c = i * 32 + lane;             // 0..511
      const int kr = c >> 4, kc8 = (c & 15) * 8;
      async_copy_b128(ko_ + (unsigned)(kr * 128 + kc8) * 2,
                      kp + (long)(kk2 + kr) * HD_ + kc8);
    }
#pragma unroll
    for (int i = 0; i < 16; ++i) {
      const int c = i * 32 + lane;             // 0..511
      const int vr = c >> 2, vc8 = (c & 3) * 8;
      async_copy_b128(vo_ + (unsigned)(vr * 32 + vc8) * 2,
                      vp + (long)vr * T_ + kk2 + vc8);
    }
  };

  v16bf qf[4];
#pragma unroll
  for (int kc = 0; kc < 4; ++kc) qf[kc] = load_frag(qp, qi0, HD_, kc * 32, lane);

  v8f o[8];
#pragma unroll
  for (int c = 0; c < 8; ++c) o[c] = (v8f){0.f,0.f,0.f,0.f,0.f,0.f,0.f,0.f};

  const float scale = 0.088388347648318447f;    // 1/sqrt(128)
  float mrun = -3.0e38f, lrun = 0.0f;
  const int m0 = (lane >> 4) * 8;
  const int n0 = lane & 15;

  stage_kv(0, 0);
  int s = 0;
  for (int kk = 0; kk < qi0 + 16; kk += 32) {
    const bool has_next = (kk + 32) < (qi0 + 16);
    if (has_next) {
      wait_ds0();              // prior ds reads of stage s^1 fully drained
      stage_kv(s ^ 1, kk + 32);
      wait_async_le32();       // current stage (issued first, in-order) resident
    } else {
      wait_async_le0();
    }

    const bf16* ks = &ksm[s][0];
    const bf16* vs = &vsm[s][0];
    v16bf kb0[4], kb1[4];
#pragma unroll
    for (int kc = 0; kc < 4; ++kc) {           // batched ds_load_b128, short latency
      kb0[kc] = load_frag(ks, 0,  128, kc * 32, lane);
      kb1[kc] = load_frag(ks, 16, 128, kc * 32, lane);
    }
    v8f s0 = (v8f){0.f,0.f,0.f,0.f,0.f,0.f,0.f,0.f};
    v8f s1 = (v8f){0.f,0.f,0.f,0.f,0.f,0.f,0.f,0.f};
#pragma unroll
    for (int kc = 0; kc < 4; ++kc) {
      s0 = wmma_bf16(qf[kc], kb0[kc], s0);
      s1 = wmma_bf16(qf[kc], kb1[kc], s1);
    }
#pragma unroll
    for (int r = 0; r < 8; ++r) {
      ssm[m0 + r][n0]      = s0[r];
      ssm[m0 + r][16 + n0] = s1[r];
    }
    __syncthreads();
    if (lane < 16) {                              // online softmax, one row/lane
      const int row = lane;
      const int qglob = qi0 + row;
      float sv[32];
      float mnew = mrun;
#pragma unroll
      for (int j = 0; j < 32; ++j) {
        float val = (kk + j <= qglob) ? ssm[row][j] * scale : -3.0e38f;
        sv[j] = val;
        mnew = fmaxf(mnew, val);
      }
      float alpha = __expf(mrun - mnew);
      float ps = 0.0f;
#pragma unroll
      for (int j = 0; j < 32; ++j) {
        float p = (sv[j] > -1.0e38f) ? __expf(sv[j] - mnew) : 0.0f;
        psm[row][j] = (bf16)p;
        ps += p;
      }
      lrun = lrun * alpha + ps;
      mrun = mnew;
      alphas[row] = alpha;
    }
    __syncthreads();
    v16bf pf = load_frag(&psm[0][0], 0, 40, 0, lane);   // P as A-fragment (16x32)
    v16bf vb[8];
#pragma unroll
    for (int c = 0; c < 8; ++c) vb[c] = load_frag(vs, c * 16, 32, 0, lane);
#pragma unroll
    for (int r = 0; r < 8; ++r) {
      float a = alphas[m0 + r];
#pragma unroll
      for (int c = 0; c < 8; ++c) o[c][r] *= a;
    }
#pragma unroll
    for (int c = 0; c < 8; ++c)                   // O += P * V
      o[c] = wmma_bf16(pf, vb[c], o[c]);
    s ^= 1;
  }
  if (lane < 16) linv[lane] = 1.0f / lrun;
  __syncthreads();
#pragma unroll
  for (int r = 0; r < 8; ++r) {
    const long trow = (long)(b * T_ + qi0 + m0 + r) * C_ + h * HD_;
    float li = linv[m0 + r];
#pragma unroll
    for (int c = 0; c < 8; ++c)
      ao[trow + c * 16 + n0] = (bf16)(o[c][r] * li);
  }
}

// ---------------------------------------------------------------- launch
extern "C" void kernel_launch(void* const* d_in, const int* in_sizes, int n_in,
                              void* d_out, int out_size, void* d_ws, size_t ws_size,
                              hipStream_t stream) {
  (void)in_sizes; (void)n_in; (void)out_size; (void)ws_size;
  const float* x    = (const float*)d_in[0];
  const float* cosp = (const float*)d_in[1];
  const float* sinp = (const float*)d_in[2];
  const float* Wqkv = (const float*)d_in[3];
  const float* Wout = (const float*)d_in[4];
  float* out = (float*)d_out;
  char* ws = (char*)d_ws;

  // workspace layout (bytes)
  bf16* xbf    = (bf16*)(ws + 0L);           //  16 MB : x bf16
  bf16* wqkvbf = (bf16*)(ws + 16777216L);    //  24 MB : Wqkv bf16
  bf16* woutbf = (bf16*)(ws + 41943040L);    //   8 MB : Wout bf16
  bf16* qkvbf  = (bf16*)(ws + 50331648L);    //  48 MB : qkv bf16
  bf16* qbf    = (bf16*)(ws + 100663296L);   //  16 MB : q roped [B,NH,T,HD]
  bf16* kbf    = (bf16*)(ws + 117440512L);   //  16 MB : k roped [B,NH,T,HD]
  bf16* vtbf   = (bf16*)(ws + 134217728L);   //  16 MB : v transposed [B,NH,HD,T]
  bf16* aobf   = (bf16*)(ws + 150994944L);   //  16 MB : attn out [B,T,C]

  const long nx  = (long)B_ * T_ * C_;       // 8388608
  const long nwq = 3L * C_ * C_;             // 12582912
  const long nwo = (long)C_ * C_;            // 4194304

  cvt_f32_bf16<<<(int)((nx  + 255) / 256), 256, 0, stream>>>(x,    xbf,    nx);
  cvt_f32_bf16<<<(int)((nwq + 255) / 256), 256, 0, stream>>>(Wqkv, wqkvbf, nwq);
  cvt_f32_bf16<<<(int)((nwo + 255) / 256), 256, 0, stream>>>(Wout, woutbf, nwo);

  // qkv = x * Wqkv^T : [4096 x 2048] x [2048 x 6144]
  gemm_tiled<bf16><<<dim3(48, 32), 256, 0, stream>>>(xbf, wqkvbf, qkvbf,
                                                     3 * C_, C_);

  rope_split<<<B_ * T_ * NH_, 64, 0, stream>>>(qkvbf, cosp, sinp, qbf, kbf, vtbf);

  attn_kernel<<<B_ * NH_ * (T_ / 16), 32, 0, stream>>>(qbf, kbf, vtbf, aobf);

  // out = ao * Wout^T : [4096 x 2048] x [2048 x 2048] -> f32
  gemm_tiled<float><<<dim3(16, 32), 256, 0, stream>>>(aobf, woutbf, out,
                                                      C_, C_);
}
